// GenScore_46737834115257
// MI455X (gfx1250) — matrix operand order repaired and checked by reading, not compiled, not run on hardware
//
#include <hip/hip_runtime.h>
#include <hip/hip_bf16.h>
#include <stdint.h>

// ---- problem constants (match reference) ----
#define BB   8
#define NL   64
#define NT   512
#define CIN  128
#define HID  256
#define KK   10
#define EE   8192
#define BN_EPS 1e-5f

// output layout (flat, in return order)
#define PI_OFF  0
#define SG_OFF  (BB*NL*NT*KK)                 // 2621440
#define MU_OFF  (2*BB*NL*NT*KK)               // 5242880
#define D_OFF   (3*BB*NL*NT*KK)               // 7864320
#define AT_OFF  (D_OFF + BB*NL*NT)            // 8126464
#define BD_OFF  (AT_OFF + (BB*NL)*17)         // 8135168

typedef __attribute__((ext_vector_type(16))) __bf16 v16bf;
typedef __attribute__((ext_vector_type(8)))  float  v8f;

__device__ __forceinline__ float eluf(float x) {
    return x > 0.f ? x : (__expf(x) - 1.f);
}

// ---------------------------------------------------------------------------
// Kernel 1: factored first-layer GEMMs with BatchNorm folded.
//   rows 0..511    -> P[r][j] = (h_l[r] . W1[0:128, j]) * s[j]
//   rows 512..4607 -> Q[r][j] = (h_t[r] . W1[128:256,j]) * s[j] + (b1*s + beta - mean*s)[j]
// ---------------------------------------------------------------------------
__global__ __launch_bounds__(256) void pq_kernel(
    const float* __restrict__ hl, const float* __restrict__ ht,
    const float* __restrict__ W1, const float* __restrict__ b1,
    const float* __restrict__ g,  const float* __restrict__ beta,
    const float* __restrict__ mean, const float* __restrict__ var,
    float* __restrict__ P, float* __restrict__ Q)
{
    int row  = blockIdx.x;               // 0..4607
    bool isQ = row >= (BB*NL);
    int r    = isQ ? row - BB*NL : row;
    const float* x = isQ ? (ht + (size_t)r*CIN) : (hl + (size_t)r*CIN);

    __shared__ float sx[CIN];
    if (threadIdx.x < CIN) sx[threadIdx.x] = x[threadIdx.x];
    __syncthreads();

    int j = threadIdx.x;                 // one output channel per thread
    const float* Wc = W1 + (isQ ? (size_t)CIN*HID : 0) + j;
    float acc = 0.f;
#pragma unroll 8
    for (int k = 0; k < CIN; ++k) acc = fmaf(sx[k], Wc[(size_t)k*HID], acc);

    float s = g[j] * rsqrtf(var[j] + BN_EPS);
    if (isQ) {
        float qb = b1[j]*s + beta[j] - mean[j]*s;
        Q[(size_t)r*HID + j] = acc*s + qb;
    } else {
        P[(size_t)r*HID + j] = acc*s;
    }
}

// ---------------------------------------------------------------------------
// Kernel 2: fused pair heads.
//   h_elu[b,l,t,:] = elu(P[b,l,:] + Q[b,t,:])          (never materialized)
//   z = h_elu @ [Wpi | Wsigma | Wmu]  (256 x 32, cols 30/31 zero-padded)
//   pi = softmax(z[0:10]); sigma = elu(z[10:20])+1.1; mu = elu(z[20:30])+1.0
// One block = (batch b, 16-wide t tile, 8-wide l group); one wave per l.
// Per wave: 8 K-chunks x 2 N-halves = 16x v_wmma_f32_16x16x32_bf16.
// Native (__bf16) casts -> hardware cvt_pk_bf16_f32 instead of manual RNE.
// ---------------------------------------------------------------------------
__global__ __launch_bounds__(256) void pair_heads_kernel(
    const float* __restrict__ P,  const float* __restrict__ Q,
    const float* __restrict__ Wpi, const float* __restrict__ bpi,
    const float* __restrict__ Wsg, const float* __restrict__ bsg,
    const float* __restrict__ Wmu, const float* __restrict__ bmu,
    const unsigned char* __restrict__ lmask,
    const unsigned char* __restrict__ tmask,
    float* __restrict__ out)
{
    __shared__ __bf16 sW[32 * HID];         // head weights [n][k], bf16
    __shared__ float  sBH[32];              // head biases
    __shared__ float  sQ[16 * HID];         // Q tile, f32
    __shared__ float  sP[8 * HID];          // 8 P rows, f32
    __shared__ float  sScr[8 * 16 * 32];    // per-wave C spill

    const int tid  = threadIdx.x;
    const int w    = tid >> 5;              // wave id = local l
    const int lane = tid & 31;
    const int grp  = lane >> 4;             // K half-group
    const int mrow = lane & 15;             // A row / B column within tile

    const int bx = blockIdx.x;              // 2048 blocks
    const int b  = bx >> 8;
    const int rm = bx & 255;
    const int t0 = (rm >> 3) * 16;
    const int l0 = (rm & 7) * 8;

    // stage packed head-weight matrix [32 n][256 k] as bf16 + biases
    for (int i = tid; i < 32 * HID; i += 256) {
        int n = i >> 8, k = i & (HID - 1);
        float v = 0.f;
        if (n < 10)      v = Wpi[k*KK + n];
        else if (n < 20) v = Wsg[k*KK + (n - 10)];
        else if (n < 30) v = Wmu[k*KK + (n - 20)];
        sW[i] = (__bf16)v;
    }
    if (tid < 32) {
        float v = 0.f;
        if (tid < 10)      v = bpi[tid];
        else if (tid < 20) v = bsg[tid - 10];
        else if (tid < 30) v = bmu[tid - 20];
        sBH[tid] = v;
    }
    // stage Q tile (16 consecutive t rows are contiguous) and P rows
    const float* Qg = Q + ((size_t)b*NT + t0) * HID;
    const float* Pg = P + ((size_t)b*NL + l0) * HID;
    __builtin_prefetch(Qg, 0, 1);
    __builtin_prefetch(Pg, 0, 1);
    for (int i = tid; i < 16 * HID; i += 256) sQ[i] = Qg[i];
    for (int i = tid; i < 8  * HID; i += 256) sP[i] = Pg[i];
    __syncthreads();

    const float* Pl = &sP[w * HID];

    v8f c0 = {}; // head outputs 0..15
    v8f c1 = {}; // head outputs 16..31

#pragma unroll
    for (int kc = 0; kc < 8; ++kc) {
        const int k0 = kc * 32;
        // --- A fragment: 16x32 bf16 tile of elu(P+Q) ---
        // lane layout (ISA 7.12.2, 16-bit A 16x32):
        //   lanes 0-15 : K k0+0..7 (elems 0-7), k0+16..23 (elems 8-15)
        //   lanes16-31 : K k0+8..15,            k0+24..31
        v16bf a;
#pragma unroll
        for (int i = 0; i < 8; ++i) {
            int kA = k0 + grp*8 + i;
            int kB = kA + 16;
            a[i]     = (__bf16)eluf(Pl[kA] + sQ[mrow*HID + kA]);
            a[i + 8] = (__bf16)eluf(Pl[kB] + sQ[mrow*HID + kB]);
        }

        // --- B fragments: 32x16 bf16; lane n holds column n,
        //     lanes 0-15 -> K k0+0..7, k0+16..23; lanes 16-31 -> K k0+8..15, k0+24..31
        v16bf bf0, bf1;
        const int kb = k0 + grp*8;
#pragma unroll
        for (int i = 0; i < 16; ++i) {
            int kk = kb + i + (i >= 8 ? 8 : 0);
            bf0[i] = sW[ mrow      * HID + kk];
            bf1[i] = sW[(16 + mrow)* HID + kk];
        }

        c0 = __builtin_amdgcn_wmma_f32_16x16x32_bf16(false, a, false, bf0,
                                                     (short)0, c0, false, false);
        c1 = __builtin_amdgcn_wmma_f32_16x16x32_bf16(false, a, false, bf1,
                                                     (short)0, c1, false, false);
    }

    // spill C (ISA layout: VGPR r, lane -> row r + 8*grp, col mrow)
    float* scr = &sScr[w * 512];
#pragma unroll
    for (int r = 0; r < 8; ++r) {
        int row = r + grp*8;
        scr[row*32 + mrow]      = c0[r];
        scr[row*32 + 16 + mrow] = c1[r];
    }
    __syncthreads();

    // epilogue: lanes 0..15 each own one t-row of this wave's tile
    if (lane < 16) {
        const int l = l0 + w;
        const int t = t0 + lane;
        const bool pm = (lmask[b*NL + l] != 0) && (tmask[b*NT + t] != 0);

        float z[30];
#pragma unroll
        for (int c = 0; c < 30; ++c) z[c] = scr[lane*32 + c] + sBH[c];

        // softmax over pi logits
        float m = z[0];
#pragma unroll
        for (int j = 1; j < KK; ++j) m = fmaxf(m, z[j]);
        float e[KK], sum = 0.f;
#pragma unroll
        for (int j = 0; j < KK; ++j) { e[j] = __expf(z[j] - m); sum += e[j]; }
        float inv = 1.f / sum;

        const size_t pbase = (((size_t)b*NL + l)*NT + t) * KK;
#pragma unroll
        for (int j = 0; j < KK; ++j)
            out[PI_OFF + pbase + j] = pm ? e[j]*inv : 0.f;
#pragma unroll
        for (int j = 0; j < KK; ++j)
            out[SG_OFF + pbase + j] = pm ? (eluf(z[10 + j]) + 1.1f) : 0.f;
#pragma unroll
        for (int j = 0; j < KK; ++j)
            out[MU_OFF + pbase + j] = pm ? (eluf(z[20 + j]) + 1.0f) : 0.f;
    }
}

// ---------------------------------------------------------------------------
// Kernel 3: pairwise distances (replicates sqrt(neg)->NaN->1e4 quirk + mask)
// ---------------------------------------------------------------------------
__global__ __launch_bounds__(256) void dist_kernel(
    const float* __restrict__ lp, const float* __restrict__ tp,
    const unsigned char* __restrict__ lm, const unsigned char* __restrict__ tm,
    float* __restrict__ out)
{
    int i = blockIdx.x * blockDim.x + threadIdx.x;
    if (i >= BB*NL*NT) return;
    int t  = i & (NT - 1);
    int bl = i / NT;
    int l  = bl & (NL - 1);
    int b  = bl / NL;

    const float* x = lp + ((size_t)b*NL + l) * 3;
    const float* y = tp + ((size_t)b*NT + t) * 3;
    float x0=x[0], x1=x[1], x2=x[2];
    float y0=y[0], y1=y[1], y2=y[2];
    float d2 = -2.f*(x0*y0 + x1*y1 + x2*y2)
             + (y0*y0 + y1*y1 + y2*y2)
             + (x0*x0 + x1*x1 + x2*x2);
    float d = sqrtf(d2);
    if (d != d) d = 10000.f;                 // nan_to_num(nan=1e4)
    bool pm = (lm[b*NL + l] != 0) && (tm[b*NT + t] != 0);
    out[D_OFF + i] = pm ? d : 0.f;
}

// ---------------------------------------------------------------------------
// Kernel 4: atom-type head (512 x 128) @ (128 x 17)
// ---------------------------------------------------------------------------
__global__ __launch_bounds__(256) void atom_kernel(
    const float* __restrict__ hl, const float* __restrict__ Wa,
    const float* __restrict__ ba, float* __restrict__ out)
{
    int i = blockIdx.x * blockDim.x + threadIdx.x;
    if (i >= BB*NL*17) return;
    int r = i / 17, j = i % 17;
    float acc = ba[j];
    const float* x = hl + (size_t)r * CIN;
#pragma unroll 8
    for (int k = 0; k < CIN; ++k) acc = fmaf(x[k], Wa[k*17 + j], acc);
    out[AT_OFF + i] = acc;
}

// ---------------------------------------------------------------------------
// Kernel 5: bond-type head, gathered pairs (8192 edges x 256) @ (256 x 4)
// ---------------------------------------------------------------------------
__global__ __launch_bounds__(256) void bond_kernel(
    const float* __restrict__ hl, const int* __restrict__ ei,
    const float* __restrict__ Wb, const float* __restrict__ bb,
    float* __restrict__ out)
{
    int e = blockIdx.x * blockDim.x + threadIdx.x;
    if (e >= EE) return;
    int s0 = ei[e], s1 = ei[EE + e];
    const float* xs = hl + (size_t)s0 * CIN;
    const float* xd = hl + (size_t)s1 * CIN;
    float a0 = bb[0], a1 = bb[1], a2 = bb[2], a3 = bb[3];
#pragma unroll 4
    for (int k = 0; k < CIN; ++k) {
        float vs = xs[k], vd = xd[k];
        const float* w0 = Wb + (size_t)k * 4;
        const float* w1 = Wb + (size_t)(CIN + k) * 4;
        a0 = fmaf(vs, w0[0], a0); a0 = fmaf(vd, w1[0], a0);
        a1 = fmaf(vs, w0[1], a1); a1 = fmaf(vd, w1[1], a1);
        a2 = fmaf(vs, w0[2], a2); a2 = fmaf(vd, w1[2], a2);
        a3 = fmaf(vs, w0[3], a3); a3 = fmaf(vd, w1[3], a3);
    }
    float* o = out + BD_OFF + (size_t)e * 4;
    o[0] = a0; o[1] = a1; o[2] = a2; o[3] = a3;
}

// ---------------------------------------------------------------------------
extern "C" void kernel_launch(void* const* d_in, const int* in_sizes, int n_in,
                              void* d_out, int out_size, void* d_ws, size_t ws_size,
                              hipStream_t stream)
{
    const float* h_l_x   = (const float*)d_in[0];
    const float* h_t_x   = (const float*)d_in[1];
    const unsigned char* l_mask = (const unsigned char*)d_in[2];
    const unsigned char* t_mask = (const unsigned char*)d_in[3];
    const float* h_l_pos = (const float*)d_in[4];
    const float* h_t_pos = (const float*)d_in[5];
    const int*   edge_ix = (const int*)d_in[6];
    const float* W1      = (const float*)d_in[7];
    const float* b1      = (const float*)d_in[8];
    const float* bn_g    = (const float*)d_in[9];
    const float* bn_b    = (const float*)d_in[10];
    const float* bn_m    = (const float*)d_in[11];
    const float* bn_v    = (const float*)d_in[12];
    const float* Wpi     = (const float*)d_in[13];
    const float* bpi     = (const float*)d_in[14];
    const float* Wsg     = (const float*)d_in[15];
    const float* bsg     = (const float*)d_in[16];
    const float* Wmu     = (const float*)d_in[17];
    const float* bmu     = (const float*)d_in[18];
    const float* Wat     = (const float*)d_in[19];
    const float* bat     = (const float*)d_in[20];
    const float* Wbd     = (const float*)d_in[21];
    const float* bbd     = (const float*)d_in[22];
    float* out = (float*)d_out;

    float* wsP = (float*)d_ws;                    // [512 ][256]
    float* wsQ = wsP + (size_t)BB*NL*HID;         // [4096][256]

    // 1) factored first-layer + BN fold
    pq_kernel<<<BB*NL + BB*NT, 256, 0, stream>>>(
        h_l_x, h_t_x, W1, b1, bn_g, bn_b, bn_m, bn_v, wsP, wsQ);

    // 2) fused WMMA pair-heads: 8 batches * 32 t-tiles * 8 l-groups
    pair_heads_kernel<<<BB*(NT/16)*(NL/8), 256, 0, stream>>>(
        wsP, wsQ, Wpi, bpi, Wsg, bsg, Wmu, bmu, l_mask, t_mask, out);

    // 3) distances
    dist_kernel<<<(BB*NL*NT + 255)/256, 256, 0, stream>>>(
        h_l_pos, h_t_pos, l_mask, t_mask, out);

    // 4) atom types
    atom_kernel<<<(BB*NL*17 + 255)/256, 256, 0, stream>>>(h_l_x, Wat, bat, out);

    // 5) bond types
    bond_kernel<<<(EE + 255)/256, 256, 0, stream>>>(h_l_x, edge_ix, Wbd, bbd, out);
}
